// GMN_embed_19335942766726
// MI455X (gfx1250) — compile-verified, matrix operand order checked by reading, not run on hardware
//
#include <hip/hip_runtime.h>
#include <hip/hip_bf16.h>

// ---------------------------------------------------------------------------
// GMN embedding for MI455X (gfx1250): edge/node MLPs via v_wmma_f32_16x16x32_f16
// msg kernel: 32 edges/block, 2x2 register tile per wave (A0/A1 x B0/B1 ->
// 4 WMMAs per operand-fetch group).
// ---------------------------------------------------------------------------

typedef __attribute__((ext_vector_type(16))) _Float16 v16h;
typedef __attribute__((ext_vector_type(8)))  _Float16 v8h;
typedef __attribute__((ext_vector_type(8)))  float    v8f;

#define N_NODES  30000
#define N_EDGES  300000
#define N_GRAPHS 256
#define F_NODE   32
#define F_EDGE   16
#define DD       128
#define HH       256

// ---- WMMA fragment helpers (ISA 7.12.2 layouts, wave32) -------------------
__device__ __forceinline__ v16h frag_a_lds(const _Float16* row, int kbase) {
    union { v16h v; v8h h[2]; } u;
    u.h[0] = *(const v8h*)(row + kbase);
    u.h[1] = *(const v8h*)(row + kbase + 16);
    return u.v;
}

__device__ __forceinline__ v16h frag_b_glb(const _Float16* __restrict__ wf,
                                           int NT, int kt, int nt, int lane) {
    const _Float16* p = wf + ((size_t)((kt * NT + nt) * 32 + lane)) * 16;
    union { v16h v; v8h h[2]; } u;
    u.h[0] = *(const v8h*)p;
    u.h[1] = *(const v8h*)(p + 8);
    return u.v;
}

// ---- weight pre-swizzle: row-major f32 [K][N] -> WMMA-B-fragment f16 ------
__global__ __launch_bounds__(256) void prep_weight_kernel(
    const float* __restrict__ W, _Float16* __restrict__ out, int K, int N) {
    int tid = blockIdx.x * 256 + threadIdx.x;
    int total = (K >> 5) * (N >> 4) * 32;
    if (tid >= total) return;
    int lane = tid & 31;
    int frag = tid >> 5;
    int NT = N >> 4;
    int kt = frag / NT;
    int nt = frag - kt * NT;
    int n  = nt * 16 + (lane & 15);
    int kh = lane >> 4;
    _Float16* o = out + (size_t)tid * 16;
#pragma unroll
    for (int v = 0; v < 8; ++v) {
#pragma unroll
        for (int hv = 0; hv < 2; ++hv) {
            int koff = ((v < 4) ? 0 : 16) + kh * 8 + (v & 3) * 2 + hv;
            o[v * 2 + hv] = (_Float16)W[(size_t)(kt * 32 + koff) * N + n];
        }
    }
}

// ---- encoders (bandwidth-trivial, scalar f32) -----------------------------
__global__ __launch_bounds__(256) void encode_nodes_kernel(
    const float* __restrict__ nf, const float* __restrict__ w,
    const float* __restrict__ b, float* __restrict__ h) {
    int idx = blockIdx.x * 256 + threadIdx.x;        // N_NODES*128
    int node = idx >> 7;
    int d = idx & 127;
    float acc = b[d];
#pragma unroll
    for (int k = 0; k < F_NODE; ++k) acc += nf[node * F_NODE + k] * w[k * DD + d];
    h[idx] = acc;
}

__global__ __launch_bounds__(256) void encode_edges_kernel(
    const float* __restrict__ ef, const float* __restrict__ w,
    const float* __restrict__ b, _Float16* __restrict__ e16) {
    int idx = blockIdx.x * 256 + threadIdx.x;        // N_EDGES*128
    int e = idx >> 7;
    int d = idx & 127;
    float acc = b[d];
#pragma unroll
    for (int k = 0; k < F_EDGE; ++k) acc += ef[e * F_EDGE + k] * w[k * DD + d];
    e16[idx] = (_Float16)acc;
}

// ---- message MLP (fwd + rev over one gathered 32-edge tile) ---------------
// GEMM1: [32x384]x[384x256] relu   GEMM2: [32x256]x[256x128]
// 2x2 register tile: A0/A1 (M-tiles) x B0/B1 (N-tiles) -> 4 WMMA per K-step.
__global__ __launch_bounds__(256) void msg_kernel(
    const float* __restrict__ h, const _Float16* __restrict__ e16,
    const int* __restrict__ from_idx, const int* __restrict__ to_idx,
    const _Float16* __restrict__ w1m, const float* __restrict__ b1m,
    const _Float16* __restrict__ w2m, const float* __restrict__ b2m,
    const _Float16* __restrict__ w1r, const float* __restrict__ b1r,
    const _Float16* __restrict__ w2r, const float* __restrict__ b2r,
    float* __restrict__ agg, float* __restrict__ ragg) {
    __shared__ __align__(16) _Float16 lds_x[32 * 384];     // 24 KB
    __shared__ __align__(16) _Float16 lds_hid[32 * 256];   // 16 KB

    const int tid  = threadIdx.x;
    const int lane = tid & 31;
    const int wave = tid >> 5;
    const int m    = lane & 15;   // row within M-tile / column of D
    const int kh   = lane >> 4;
    const int edge0 = blockIdx.x * 32;

    // stage concat[hf | ht | e] for 32 edges (gather + f32->f16)
    for (int c = tid; c < 32 * 384; c += 256) {
        int row = c / 384;
        int k   = c - row * 384;
        int e   = edge0 + row;
        _Float16 v;
        if (k < 128)       v = (_Float16)h[(size_t)from_idx[e] * DD + k];
        else if (k < 256)  v = (_Float16)h[(size_t)to_idx[e] * DD + (k - 128)];
        else               v = e16[(size_t)e * DD + (k - 256)];
        lds_x[c] = v;
    }
    __syncthreads();

    for (int pass = 0; pass < 2; ++pass) {
        const _Float16* w1 = pass ? w1r : w1m;
        const _Float16* w2 = pass ? w2r : w2m;
        const float* b1 = pass ? b1r : b1m;
        const float* b2 = pass ? b2r : b2m;
        const int* sidx = pass ? from_idx : to_idx;
        float* dst = pass ? ragg : agg;

        // GEMM1: 8 waves x (2 N-tiles x 2 M-tiles) -> H=256
        {
            v8f acc[2][2] = {};   // [nn][mtile]
#pragma unroll
            for (int kt = 0; kt < 12; ++kt) {
                // rev pass swaps hf<->ht blocks: remap K-tile (swap at 128 gran.)
                int kts = pass ? ((kt < 4) ? kt + 4 : (kt < 8) ? kt - 4 : kt) : kt;
                int kb = kts * 32 + kh * 8;
                v16h a0 = frag_a_lds(&lds_x[m * 384], kb);
                v16h a1 = frag_a_lds(&lds_x[(m + 16) * 384], kb);
                v16h b0 = frag_b_glb(w1, 16, kt, wave * 2 + 0, lane);
                v16h b1 = frag_b_glb(w1, 16, kt, wave * 2 + 1, lane);
                acc[0][0] = __builtin_amdgcn_wmma_f32_16x16x32_f16(
                                false, a0, false, b0, (short)0, acc[0][0], false, false);
                acc[0][1] = __builtin_amdgcn_wmma_f32_16x16x32_f16(
                                false, a1, false, b0, (short)0, acc[0][1], false, false);
                acc[1][0] = __builtin_amdgcn_wmma_f32_16x16x32_f16(
                                false, a0, false, b1, (short)0, acc[1][0], false, false);
                acc[1][1] = __builtin_amdgcn_wmma_f32_16x16x32_f16(
                                false, a1, false, b1, (short)0, acc[1][1], false, false);
            }
#pragma unroll
            for (int nn = 0; nn < 2; ++nn) {
                int nt = wave * 2 + nn;
                float bias = b1[nt * 16 + m];
#pragma unroll
                for (int r = 0; r < 8; ++r) {
                    int mm = kh * 8 + r;             // D layout: M = r + 8*(lane>>4)
                    float v0 = acc[nn][0][r] + bias;
                    float v1 = acc[nn][1][r] + bias;
                    v0 = v0 > 0.f ? v0 : 0.f;
                    v1 = v1 > 0.f ? v1 : 0.f;
                    lds_hid[mm * 256 + nt * 16 + m]        = (_Float16)v0;
                    lds_hid[(mm + 16) * 256 + nt * 16 + m] = (_Float16)v1;
                }
            }
        }
        __syncthreads();

        // GEMM2: 8 waves x 1 N-tile x 2 M-tiles -> D=128
        {
            int nt = wave;
            v8f c0 = {};
            v8f c1 = {};
#pragma unroll
            for (int kt = 0; kt < 8; ++kt) {
                int kb = kt * 32 + kh * 8;
                v16h b  = frag_b_glb(w2, 8, kt, nt, lane);
                v16h a0 = frag_a_lds(&lds_hid[m * 256], kb);
                v16h a1 = frag_a_lds(&lds_hid[(m + 16) * 256], kb);
                c0 = __builtin_amdgcn_wmma_f32_16x16x32_f16(
                         false, a0, false, b, (short)0, c0, false, false);
                c1 = __builtin_amdgcn_wmma_f32_16x16x32_f16(
                         false, a1, false, b, (short)0, c1, false, false);
            }
            float bias = b2[nt * 16 + m];
            int n = nt * 16 + m;
#pragma unroll
            for (int r = 0; r < 8; ++r) {
                int mm = kh * 8 + r;
                int e0 = edge0 + mm;
                int e1 = edge0 + 16 + mm;
                atomicAdd(&dst[(size_t)sidx[e0] * DD + n], c0[r] + bias);
                atomicAdd(&dst[(size_t)sidx[e1] * DD + n], c1[r] + bias);
            }
        }
        __syncthreads();   // protect lds_hid before rev pass rewrites it
    }
}

// ---- node update MLP: h += MLP2(concat[agg, ragg, h]) ---------------------
__global__ __launch_bounds__(256) void upd_kernel(
    float* __restrict__ h, const float* __restrict__ agg, const float* __restrict__ ragg,
    const _Float16* __restrict__ w1, const float* __restrict__ b1,
    const _Float16* __restrict__ w2, const float* __restrict__ b2) {
    __shared__ __align__(16) _Float16 lds_x[16 * 384];
    __shared__ __align__(16) _Float16 lds_hid[16 * 256];

    const int tid  = threadIdx.x;
    const int lane = tid & 31;
    const int wave = tid >> 5;
    const int m    = lane & 15;
    const int kh   = lane >> 4;
    const int node0 = blockIdx.x * 16;

    for (int c = tid; c < 16 * 384; c += 256) {
        int row = c / 384;
        int k   = c - row * 384;
        size_t nbase = (size_t)(node0 + row) * DD;
        float v;
        if (k < 128)       v = agg[nbase + k];
        else if (k < 256)  v = ragg[nbase + (k - 128)];
        else               v = h[nbase + (k - 256)];
        lds_x[c] = (_Float16)v;
    }
    __syncthreads();

    {
        v8f acc[2] = {};   // 2 N-tiles share the A fragment
#pragma unroll
        for (int kt = 0; kt < 12; ++kt) {
            int kb = kt * 32 + kh * 8;
            v16h a  = frag_a_lds(&lds_x[m * 384], kb);
            v16h b0 = frag_b_glb(w1, 16, kt, wave * 2 + 0, lane);
            v16h b1 = frag_b_glb(w1, 16, kt, wave * 2 + 1, lane);
            acc[0] = __builtin_amdgcn_wmma_f32_16x16x32_f16(
                         false, a, false, b0, (short)0, acc[0], false, false);
            acc[1] = __builtin_amdgcn_wmma_f32_16x16x32_f16(
                         false, a, false, b1, (short)0, acc[1], false, false);
        }
#pragma unroll
        for (int nn = 0; nn < 2; ++nn) {
            int nt = wave * 2 + nn;
            float bias = b1[nt * 16 + m];
#pragma unroll
            for (int r = 0; r < 8; ++r) {
                int mm = kh * 8 + r;
                float v = acc[nn][r] + bias;
                v = v > 0.f ? v : 0.f;
                lds_hid[mm * 256 + nt * 16 + m] = (_Float16)v;
            }
        }
    }
    __syncthreads();

    {
        int nt = wave;
        v8f c = {};
#pragma unroll
        for (int kt = 0; kt < 8; ++kt) {
            int kb = kt * 32 + kh * 8;
            v16h a = frag_a_lds(&lds_hid[m * 256], kb);
            v16h b = frag_b_glb(w2, 8, kt, nt, lane);
            c = __builtin_amdgcn_wmma_f32_16x16x32_f16(
                    false, a, false, b, (short)0, c, false, false);
        }
        float bias = b2[nt * 16 + m];
        int n = nt * 16 + m;
#pragma unroll
        for (int r = 0; r < 8; ++r) {
            int mm = kh * 8 + r;
            h[(size_t)(node0 + mm) * DD + n] += c[r] + bias;   // residual
        }
    }
}

// ---- gated readout: g = h@agg_w1+b1; sigmoid(g[:128])*g[128:]; segsum -----
__global__ __launch_bounds__(256) void gated_kernel(
    const float* __restrict__ h, const _Float16* __restrict__ w1,
    const float* __restrict__ b1, const int* __restrict__ graph_idx,
    float* __restrict__ gv) {
    __shared__ __align__(16) _Float16 lds_x[16 * 128];
    __shared__ float lds_g[16 * 256];

    const int tid  = threadIdx.x;
    const int lane = tid & 31;
    const int wave = tid >> 5;
    const int m    = lane & 15;
    const int kh   = lane >> 4;
    const int node0 = blockIdx.x * 16;

    for (int c = tid; c < 16 * 128; c += 256) {
        int row = c >> 7;
        int k   = c & 127;
        lds_x[c] = (_Float16)h[(size_t)(node0 + row) * DD + k];
    }
    __syncthreads();

    {
        v8f acc[2] = {};
#pragma unroll
        for (int kt = 0; kt < 4; ++kt) {
            int kb = kt * 32 + kh * 8;
            v16h a  = frag_a_lds(&lds_x[m * 128], kb);
            v16h b0 = frag_b_glb(w1, 16, kt, wave * 2 + 0, lane);
            v16h b1 = frag_b_glb(w1, 16, kt, wave * 2 + 1, lane);
            acc[0] = __builtin_amdgcn_wmma_f32_16x16x32_f16(
                         false, a, false, b0, (short)0, acc[0], false, false);
            acc[1] = __builtin_amdgcn_wmma_f32_16x16x32_f16(
                         false, a, false, b1, (short)0, acc[1], false, false);
        }
#pragma unroll
        for (int nn = 0; nn < 2; ++nn) {
            int nt = wave * 2 + nn;
            float bias = b1[nt * 16 + m];
#pragma unroll
            for (int r = 0; r < 8; ++r) {
                int mm = kh * 8 + r;
                lds_g[mm * 256 + nt * 16 + m] = acc[nn][r] + bias;
            }
        }
    }
    __syncthreads();

    for (int c = tid; c < 16 * 128; c += 256) {
        int row = c >> 7;
        int n   = c & 127;
        float gl = lds_g[row * 256 + n];
        float gr = lds_g[row * 256 + 128 + n];
        float val = gr / (1.f + __expf(-gl));        // sigmoid(gl) * gr
        int node = node0 + row;
        atomicAdd(&gv[(size_t)graph_idx[node] * DD + n], val);
    }
}

// ---- final: gv@agg_w2+b2, pair squared distance ---------------------------
__global__ __launch_bounds__(128) void final_kernel(
    const float* __restrict__ gv, const float* __restrict__ w2,
    const float* __restrict__ b2, float* __restrict__ out) {
    __shared__ float red[128];
    int p = blockIdx.x;          // 128 pairs
    int n = threadIdx.x;
    const float* gx = gv + (size_t)(2 * p) * DD;
    const float* gy = gv + (size_t)(2 * p + 1) * DD;
    float x = b2[n], y = b2[n];
    for (int k = 0; k < DD; ++k) {
        float w = w2[k * DD + n];
        x += gx[k] * w;
        y += gy[k] * w;
    }
    float d = x - y;
    red[n] = d * d;
    __syncthreads();
    for (int s = 64; s > 0; s >>= 1) {
        if (n < s) red[n] += red[n + s];
        __syncthreads();
    }
    if (n == 0) out[p] = -red[0];
}

// ---------------------------------------------------------------------------
extern "C" void kernel_launch(void* const* d_in, const int* in_sizes, int n_in,
                              void* d_out, int out_size, void* d_ws, size_t ws_size,
                              hipStream_t stream) {
    (void)in_sizes; (void)n_in; (void)out_size; (void)ws_size;
    const float* node_features = (const float*)d_in[0];
    const float* edge_features = (const float*)d_in[1];
    const int*   from_idx      = (const int*)d_in[2];
    const int*   to_idx        = (const int*)d_in[3];
    const int*   graph_idx     = (const int*)d_in[4];
    const float* enc_wn = (const float*)d_in[5];
    const float* enc_bn = (const float*)d_in[6];
    const float* enc_we = (const float*)d_in[7];
    const float* enc_be = (const float*)d_in[8];
    const float* msg_w1 = (const float*)d_in[9];
    const float* msg_b1 = (const float*)d_in[10];
    const float* msg_w2 = (const float*)d_in[11];
    const float* msg_b2 = (const float*)d_in[12];
    const float* rmsg_w1 = (const float*)d_in[13];
    const float* rmsg_b1 = (const float*)d_in[14];
    const float* rmsg_w2 = (const float*)d_in[15];
    const float* rmsg_b2 = (const float*)d_in[16];
    const float* upd_w1 = (const float*)d_in[17];
    const float* upd_b1 = (const float*)d_in[18];
    const float* upd_w2 = (const float*)d_in[19];
    const float* upd_b2 = (const float*)d_in[20];
    const float* agg_w1 = (const float*)d_in[21];
    const float* agg_b1 = (const float*)d_in[22];
    const float* agg_w2 = (const float*)d_in[23];
    const float* agg_b2 = (const float*)d_in[24];

    char* ws = (char*)d_ws;
    size_t off = 0;
    auto alloc = [&](size_t bytes) -> void* {
        void* p = ws + off;
        off += (bytes + 255) & ~(size_t)255;
        return p;
    };

    float* hbuf = (float*)alloc((size_t)N_NODES * DD * sizeof(float));
    float* agg  = (float*)alloc((size_t)N_NODES * DD * sizeof(float));
    float* ragg = (float*)alloc((size_t)N_NODES * DD * sizeof(float));
    float* gv   = (float*)alloc((size_t)N_GRAPHS * DD * sizeof(float));
    _Float16* e16 = (_Float16*)alloc((size_t)N_EDGES * DD * sizeof(_Float16));
    _Float16* w1m = (_Float16*)alloc((size_t)384 * 256 * sizeof(_Float16));
    _Float16* w1r = (_Float16*)alloc((size_t)384 * 256 * sizeof(_Float16));
    _Float16* w1u = (_Float16*)alloc((size_t)384 * 256 * sizeof(_Float16));
    _Float16* w2m = (_Float16*)alloc((size_t)256 * 128 * sizeof(_Float16));
    _Float16* w2r = (_Float16*)alloc((size_t)256 * 128 * sizeof(_Float16));
    _Float16* w2u = (_Float16*)alloc((size_t)256 * 128 * sizeof(_Float16));
    _Float16* w1g = (_Float16*)alloc((size_t)128 * 256 * sizeof(_Float16));

    // weight pre-swizzle (B fragments, f16)
    prep_weight_kernel<<<(384 * 256 / 16 + 255) / 256, 256, 0, stream>>>(msg_w1, w1m, 384, 256);
    prep_weight_kernel<<<(384 * 256 / 16 + 255) / 256, 256, 0, stream>>>(rmsg_w1, w1r, 384, 256);
    prep_weight_kernel<<<(384 * 256 / 16 + 255) / 256, 256, 0, stream>>>(upd_w1, w1u, 384, 256);
    prep_weight_kernel<<<(256 * 128 / 16 + 255) / 256, 256, 0, stream>>>(msg_w2, w2m, 256, 128);
    prep_weight_kernel<<<(256 * 128 / 16 + 255) / 256, 256, 0, stream>>>(rmsg_w2, w2r, 256, 128);
    prep_weight_kernel<<<(256 * 128 / 16 + 255) / 256, 256, 0, stream>>>(upd_w2, w2u, 256, 128);
    prep_weight_kernel<<<(128 * 256 / 16 + 255) / 256, 256, 0, stream>>>(agg_w1, w1g, 128, 256);

    // encoders
    encode_nodes_kernel<<<N_NODES * DD / 256, 256, 0, stream>>>(node_features, enc_wn, enc_bn, hbuf);
    encode_edges_kernel<<<N_EDGES * DD / 256, 256, 0, stream>>>(edge_features, enc_we, enc_be, e16);

    // message-passing layers
    for (int layer = 0; layer < 3; ++layer) {
        hipMemsetAsync(agg,  0, (size_t)N_NODES * DD * sizeof(float), stream);
        hipMemsetAsync(ragg, 0, (size_t)N_NODES * DD * sizeof(float), stream);
        msg_kernel<<<N_EDGES / 32, 256, 0, stream>>>(
            hbuf, e16, from_idx, to_idx,
            w1m, msg_b1, w2m, msg_b2,
            w1r, rmsg_b1, w2r, rmsg_b2,
            agg, ragg);
        upd_kernel<<<N_NODES / 16, 256, 0, stream>>>(
            hbuf, agg, ragg, w1u, upd_b1, w2u, upd_b2);
    }

    // gated graph readout + pair distance
    hipMemsetAsync(gv, 0, (size_t)N_GRAPHS * DD * sizeof(float), stream);
    gated_kernel<<<N_NODES / 16, 256, 0, stream>>>(hbuf, w1g, agg_b1, graph_idx, gv);
    final_kernel<<<N_GRAPHS / 2, 128, 0, stream>>>(gv, agg_w2, agg_b2, (float*)d_out);
}
